// improvedDeformableLocalGraphAttention_14826227106564
// MI455X (gfx1250) — matrix operand order repaired and checked by reading, not compiled, not run on hardware
//
#include <hip/hip_runtime.h>

typedef __attribute__((ext_vector_type(16))) __bf16 v16bf;
typedef __attribute__((ext_vector_type(8)))  __bf16 v8bf;
typedef __attribute__((ext_vector_type(8)))  float  v8f;

#define CDIM  384
#define NPTS  2048
#define BATCH 4
#define KNN   10
#define BM    128
#define BN    128
#define BK    32
#define LDT   40   // padded LDS row stride in bf16 elems (80B: 16B-aligned, bank-spread)
#define TILE_BYTES (BM * LDT * 2)   // 10240 B per buffer

union FragBF { v16bf v; v8bf h[2]; };

__device__ __forceinline__ float wave_sum(float v) {
#pragma unroll
  for (int o = 16; o > 0; o >>= 1) v += __shfl_xor(v, o, 32);
  return v;
}

// ---------------- prep: f32 -> bf16 conversions ----------------
__global__ __launch_bounds__(256)
void cvt_bf16_kernel(const float* __restrict__ src, __bf16* __restrict__ dst) {
  const long t = (long)blockIdx.x * 256 + threadIdx.x;
  dst[t] = (__bf16)src[t];
}

// dst[n][k] = srcA[k][n] (- srcSub[k][n]) ; 384x384 transpose+convert
__global__ __launch_bounds__(256)
void prep_weight_kernel(const float* __restrict__ srcA, const float* __restrict__ srcSub,
                        __bf16* __restrict__ dst) {
  const int t = blockIdx.x * 256 + threadIdx.x;
  const int n = t / CDIM, k = t % CDIM;
  float v = srcA[k * CDIM + n];
  if (srcSub) v -= srcSub[k * CDIM + n];
  dst[n * CDIM + k] = (__bf16)v;
}

// ---------------- KNN top-10 (self-attn) + per-query scale ----------------
__global__ __launch_bounds__(256)
void knn_kernel(const float* __restrict__ qpos, int* __restrict__ idx,
                float* __restrict__ scale) {
  __shared__ float sp[NPTS * 3];
  const int b  = blockIdx.x >> 3;
  const int q0 = (blockIdx.x & 7) * 256;
  const float* base = qpos + (long)b * NPTS * 3;
  for (int i = threadIdx.x; i < NPTS * 3; i += 256) sp[i] = base[i];
  __syncthreads();
  const int qn = q0 + threadIdx.x;
  const float qx = sp[qn*3+0], qy = sp[qn*3+1], qz = sp[qn*3+2];
  float bd[KNN]; int bi[KNN];
#pragma unroll
  for (int j = 0; j < KNN; ++j) { bd[j] = 3.0e38f; bi[j] = 0; }
  for (int j = 0; j < NPTS; ++j) {
    const float dx = sp[j*3+0]-qx, dy = sp[j*3+1]-qy, dz = sp[j*3+2]-qz;
    float cd = dx*dx + dy*dy + dz*dz;
    int ci = j;
    if (cd < bd[KNN-1]) {
#pragma unroll
      for (int p = 0; p < KNN; ++p)
        if (cd < bd[p]) { float td=bd[p]; int ti=bi[p]; bd[p]=cd; bi[p]=ci; cd=td; ci=ti; }
    }
  }
  float mn0=3e38f,mn1=3e38f,mn2=3e38f, mx0=-3e38f,mx1=-3e38f,mx2=-3e38f;
#pragma unroll
  for (int j = 0; j < KNN; ++j) {
    const int p = bi[j];
    const float x = sp[p*3+0], y = sp[p*3+1], z = sp[p*3+2];
    mn0 = fminf(mn0,x); mx0 = fmaxf(mx0,x);
    mn1 = fminf(mn1,y); mx1 = fmaxf(mx1,y);
    mn2 = fminf(mn2,z); mx2 = fmaxf(mx2,z);
  }
  const long g = (long)b * NPTS + qn;
#pragma unroll
  for (int j = 0; j < KNN; ++j) idx[g*KNN + j] = bi[j];
  scale[g*3+0] = (mx0-mn0)*0.5f;
  scale[g*3+1] = (mx1-mn1)*0.5f;
  scale[g*3+2] = (mx2-mn2)*0.5f;
}

// ---------------- bf16 WMMA GEMM: C[M,384] = A[M or gather,384] @ Bt^T (+bias) ----------------
// Bt is pre-transposed [N=384][K=384]. Optional row gather (per-batch indices).
// Tiles are streamed into LDS with GLOBAL_LOAD_ASYNC_TO_LDS_B128 (ASYNCcnt), double-buffered.
__global__ __launch_bounds__(256)
void gemm384_bf16_kernel(const __bf16* __restrict__ A,
                         const int* __restrict__ gidx, int rows_per_batch, int batch_stride,
                         const __bf16* __restrict__ Bt,
                         const float* __restrict__ bias,
                         float* __restrict__ Cf, __bf16* __restrict__ Cb) {
  __shared__ alignas(16) __bf16 sA[2][BM * LDT];
  __shared__ alignas(16) __bf16 sB[2][BN * LDT];
  const int tid  = threadIdx.x;
  const int lane = tid & 31;
  const int wave = tid >> 5;
  const int wm = wave & 3;          // 4 wave-rows of 32
  const int wn = wave >> 2;         // 2 wave-cols of 64
  const long m0 = (long)blockIdx.x * BM;
  const int  n0 = blockIdx.y * BN;

  // ---- hoisted per-thread staging addresses (2 x 16B chunks each for A and B) ----
  unsigned long aAddr[2], bAddr[2];
  unsigned int  aLds[2], bLds[2];
#pragma unroll
  for (int i = 0; i < 2; ++i) {
    const int c = tid + i * 256;          // chunk id in [0,512)
    const int row = c >> 2, part = c & 3; // 4 x 16B chunks per 32-elem row
    const long gm = m0 + row;
    long srow = gm;
    if (gidx) srow = (gm / rows_per_batch) * batch_stride + gidx[gm];
    aAddr[i] = (unsigned long)(A  + srow * CDIM + part * 8);
    bAddr[i] = (unsigned long)(Bt + (long)(n0 + row) * CDIM + part * 8);
    aLds[i]  = (unsigned int)(unsigned long)&sA[0][row * LDT + part * 8];
    bLds[i]  = (unsigned int)(unsigned long)&sB[0][row * LDT + part * 8];
  }

  v8f acc[2][4];
#pragma unroll
  for (int i = 0; i < 2; ++i)
#pragma unroll
    for (int j = 0; j < 4; ++j) acc[i][j] = {};

  // wave32 fragment addressing (per CDNA5 ISA layouts)
  const int arow = wm * 32 + (lane & 15);
  const int ako  = (lane >> 4) * 8;     // A: two 8-elem K chunks at ako, ako+16
  const int brow = wn * 64 + (lane & 15);
  const int bko  = (lane >> 4) * 8;     // B: two 8-elem K chunks at bko, bko+16

  // ---- prologue: async-fill buffer 0 with tile k0=0 ----
#pragma unroll
  for (int i = 0; i < 2; ++i) {
    asm volatile("global_load_async_to_lds_b128 %0, %1, off"
                 :: "v"(aLds[i]), "v"(aAddr[i]) : "memory");
    asm volatile("global_load_async_to_lds_b128 %0, %1, off"
                 :: "v"(bLds[i]), "v"(bAddr[i]) : "memory");
  }
  asm volatile("s_wait_asynccnt 0x0" ::: "memory");
  __syncthreads();

#pragma unroll
  for (int t = 0; t < CDIM / BK; ++t) {
    const int buf = t & 1;
    // issue next tile into the other buffer while this one is consumed
    if (t < CDIM / BK - 1) {
      const unsigned long koff = (unsigned long)((t + 1) * BK) * 2;  // bytes
      const unsigned int  loff = (unsigned int)((buf ^ 1) * TILE_BYTES);
#pragma unroll
      for (int i = 0; i < 2; ++i) {
        asm volatile("global_load_async_to_lds_b128 %0, %1, off"
                     :: "v"(aLds[i] + loff), "v"(aAddr[i] + koff) : "memory");
        asm volatile("global_load_async_to_lds_b128 %0, %1, off"
                     :: "v"(bLds[i] + loff), "v"(bAddr[i] + koff) : "memory");
      }
    }

    FragBF fa[2], fb[4];
#pragma unroll
    for (int tm = 0; tm < 2; ++tm) {
      const __bf16* p = &sA[buf][(arow + tm * 16) * LDT + ako];
      fa[tm].h[0] = *(const v8bf*)p;
      fa[tm].h[1] = *(const v8bf*)(p + 16);
    }
#pragma unroll
    for (int tn = 0; tn < 4; ++tn) {
      const __bf16* p = &sB[buf][(brow + tn * 16) * LDT + bko];
      fb[tn].h[0] = *(const v8bf*)p;
      fb[tn].h[1] = *(const v8bf*)(p + 16);
    }
#pragma unroll
    for (int tm = 0; tm < 2; ++tm)
#pragma unroll
      for (int tn = 0; tn < 4; ++tn)
        acc[tm][tn] = __builtin_amdgcn_wmma_f32_16x16x32_bf16(
            false, fa[tm].v, false, fb[tn].v, (short)0, acc[tm][tn], false, false);

    // next buffer's async fill complete + visible to all waves before next iter
    asm volatile("s_wait_asynccnt 0x0" ::: "memory");
    __syncthreads();
  }

  const int rbase = wm * 32 + ((lane >> 4) * 8);
  const int cbase = n0 + wn * 64 + (lane & 15);
#pragma unroll
  for (int tm = 0; tm < 2; ++tm) {
#pragma unroll
    for (int tn = 0; tn < 4; ++tn) {
      const int col = cbase + tn * 16;
      const float badd = bias ? bias[col] : 0.f;
      const long row0 = m0 + rbase + tm * 16;
#pragma unroll
      for (int r = 0; r < 8; ++r) {
        const float v = acc[tm][tn][r] + badd;
        const long off = (row0 + r) * CDIM + col;
        if (Cf) Cf[off] = v; else Cb[off] = (__bf16)v;
      }
    }
  }
}

// ---------------- LayerNorm + GELU + W2 + tanh + deformed positions ----------------
__global__ __launch_bounds__(256)
void ln_gelu_offset_kernel(const float* __restrict__ G, const float* __restrict__ qW1b,
                           const float* __restrict__ ln_g, const float* __restrict__ ln_b,
                           const float* __restrict__ W2, const float* __restrict__ qpos,
                           const int* __restrict__ idx, const float* __restrict__ scale,
                           float* __restrict__ shiftpos) {
  const int lane = threadIdx.x & 31;
  const long m = (long)blockIdx.x * 8 + (threadIdx.x >> 5);   // row in [0,81920)
  const long n = m / KNN;
  const int  b = (int)(m / (NPTS * KNN));
  float h[12];
#pragma unroll
  for (int i = 0; i < 12; ++i) {
    const int c = lane + 32 * i;
    h[i] = G[m * CDIM + c] + qW1b[n * CDIM + c];
  }
  float s = 0.f;
#pragma unroll
  for (int i = 0; i < 12; ++i) s += h[i];
  const float mu = wave_sum(s) * (1.0f / CDIM);
  float s2 = 0.f;
#pragma unroll
  for (int i = 0; i < 12; ++i) { const float d = h[i] - mu; s2 += d * d; }
  const float rstd = rsqrtf(wave_sum(s2) * (1.0f / CDIM) + 1e-5f);
  float a0 = 0.f, a1 = 0.f, a2 = 0.f;
#pragma unroll
  for (int i = 0; i < 12; ++i) {
    const int c = lane + 32 * i;
    float x = (h[i] - mu) * rstd * ln_g[c] + ln_b[c];
    x = 0.5f * x * (1.0f + erff(x * 0.70710678118654752f));   // exact GELU
    a0 += x * W2[c*3+0]; a1 += x * W2[c*3+1]; a2 += x * W2[c*3+2];
  }
  a0 = wave_sum(a0); a1 = wave_sum(a1); a2 = wave_sum(a2);
  if (lane < 3) {
    const float off = tanhf(lane == 0 ? a0 : (lane == 1 ? a1 : a2));
    const int p = idx[m];
    const float pos = qpos[((long)b * NPTS + p) * 3 + lane];
    shiftpos[m * 3 + lane] = pos + off * scale[n * 3 + lane];
  }
}

// ---------------- three_nn + inverse-distance weights ----------------
__global__ __launch_bounds__(256)
void three_nn_kernel(const float* __restrict__ qpos, const float* __restrict__ shiftpos,
                     int* __restrict__ idx3, float* __restrict__ w3) {
  __shared__ float sp[NPTS * 3];
  const int b  = blockIdx.x / 80;
  const int t0 = (blockIdx.x % 80) * 256;
  const float* base = qpos + (long)b * NPTS * 3;
  for (int i = threadIdx.x; i < NPTS * 3; i += 256) sp[i] = base[i];
  __syncthreads();
  const long m = (long)b * (NPTS * KNN) + t0 + threadIdx.x;
  const float qx = shiftpos[m*3+0], qy = shiftpos[m*3+1], qz = shiftpos[m*3+2];
  float bd[3] = {3e38f, 3e38f, 3e38f}; int bi[3] = {0, 0, 0};
  for (int j = 0; j < NPTS; ++j) {
    const float dx = sp[j*3]-qx, dy = sp[j*3+1]-qy, dz = sp[j*3+2]-qz;
    float cd = dx*dx + dy*dy + dz*dz; int ci = j;
    if (cd < bd[2]) {
#pragma unroll
      for (int p = 0; p < 3; ++p)
        if (cd < bd[p]) { float td=bd[p]; int ti=bi[p]; bd[p]=cd; bi[p]=ci; cd=td; ci=ti; }
    }
  }
  const float r0 = 1.f/(bd[0]+1e-8f), r1 = 1.f/(bd[1]+1e-8f), r2 = 1.f/(bd[2]+1e-8f);
  const float rs = 1.f / (r0 + r1 + r2);
  idx3[m*3+0] = bi[0]; idx3[m*3+1] = bi[1]; idx3[m*3+2] = bi[2];
  w3[m*3+0] = r0*rs;   w3[m*3+1] = r1*rs;   w3[m*3+2] = r2*rs;
}

// ---------------- weighted interpolation -> bf16 GEMM A-matrix ----------------
__global__ __launch_bounds__(256)
void interp_kernel(const float* __restrict__ q, const int* __restrict__ idx3,
                   const float* __restrict__ w3, __bf16* __restrict__ A2) {
  const int lane = threadIdx.x & 31;
  const long m = (long)blockIdx.x * 8 + (threadIdx.x >> 5);
  const int b = (int)(m / (NPTS * KNN));
  const float w0 = w3[m*3+0], w1 = w3[m*3+1], w2 = w3[m*3+2];
  const float* q0 = q + ((long)b * NPTS + idx3[m*3+0]) * CDIM;
  const float* q1 = q + ((long)b * NPTS + idx3[m*3+1]) * CDIM;
  const float* q2 = q + ((long)b * NPTS + idx3[m*3+2]) * CDIM;
#pragma unroll
  for (int i = 0; i < 12; ++i) {
    const int c = lane + 32 * i;
    A2[m * CDIM + c] = (__bf16)(w0*q0[c] + w1*q1[c] + w2*q2[c]);
  }
}

// ---------------- leaky-relu + max over K ----------------
__global__ __launch_bounds__(256)
void reduce_max_kernel(const float* __restrict__ P, const float* __restrict__ qWkd,
                       float* __restrict__ out) {
  const long t = (long)blockIdx.x * 256 + threadIdx.x;   // flat (n, c)
  const long n = t / CDIM;
  const int  c = (int)(t % CDIM);
  const float base = qWkd[n * CDIM + c];
  float mx = -3e38f;
#pragma unroll
  for (int k = 0; k < KNN; ++k) {
    float v = P[(n * KNN + k) * CDIM + c] + base;
    v = v > 0.f ? v : 0.2f * v;
    mx = fmaxf(mx, v);
  }
  out[t] = mx;
}

extern "C" void kernel_launch(void* const* d_in, const int* in_sizes, int n_in,
                              void* d_out, int out_size, void* d_ws, size_t ws_size,
                              hipStream_t stream) {
  (void)in_sizes; (void)n_in; (void)out_size; (void)ws_size;
  const float* q    = (const float*)d_in[0];
  const float* qpos = (const float*)d_in[1];
  const float* Wv   = (const float*)d_in[2];
  const float* bv   = (const float*)d_in[3];
  const float* W1   = (const float*)d_in[4];
  const float* b1   = (const float*)d_in[5];
  const float* ln_g = (const float*)d_in[6];
  const float* ln_b = (const float*)d_in[7];
  const float* W2   = (const float*)d_in[8];
  const float* Wk   = (const float*)d_in[9];
  const float* bk   = (const float*)d_in[10];
  float* out = (float*)d_out;

  char* w = (char*)d_ws;
  auto alloc = [&](size_t bytes) { char* p = w; w += (bytes + 255) & ~(size_t)255; return p; };

  const long NR = (long)BATCH * NPTS;    // 8192 query rows
  const long M2 = NR * KNN;              // 81920 neighbor rows
  __bf16* qbf  = (__bf16*)alloc(NR * CDIM * 2);
  __bf16* voff = (__bf16*)alloc(NR * CDIM * 2);
  __bf16* WvT  = (__bf16*)alloc(CDIM * CDIM * 2);
  __bf16* W1tT = (__bf16*)alloc(CDIM * CDIM * 2);   // W1 top half, transposed
  __bf16* W1bT = (__bf16*)alloc(CDIM * CDIM * 2);   // W1 bottom half, transposed
  __bf16* WktT = (__bf16*)alloc(CDIM * CDIM * 2);   // Wk top half, transposed
  __bf16* WkdT = (__bf16*)alloc(CDIM * CDIM * 2);   // (Wk_bot - Wk_top), transposed
  int*    idx  = (int*)  alloc(M2 * 4);
  float*  scl  = (float*)alloc(NR * 3 * 4);
  float*  qW1b = (float*)alloc(NR * CDIM * 4);
  float*  qWkd = (float*)alloc(NR * CDIM * 4);
  float*  G    = (float*)alloc(M2 * CDIM * 4);      // reused as P for second GEMM
  float*  spos = (float*)alloc(M2 * 3 * 4);
  int*    idx3 = (int*)  alloc(M2 * 3 * 4);
  float*  w3   = (float*)alloc(M2 * 3 * 4);
  __bf16* A2   = (__bf16*)alloc(M2 * CDIM * 2);

  // prep: conversions + weight transposes (split-concat algebra: [X,q]@W = X@Wtop + q@Wbot)
  cvt_bf16_kernel<<<(int)((NR * CDIM) / 256), 256, 0, stream>>>(q, qbf);
  const int wg = (CDIM * CDIM) / 256;
  prep_weight_kernel<<<wg, 256, 0, stream>>>(Wv, nullptr, WvT);
  prep_weight_kernel<<<wg, 256, 0, stream>>>(W1, nullptr, W1tT);
  prep_weight_kernel<<<wg, 256, 0, stream>>>(W1 + CDIM * CDIM, nullptr, W1bT);
  prep_weight_kernel<<<wg, 256, 0, stream>>>(Wk, nullptr, WktT);
  prep_weight_kernel<<<wg, 256, 0, stream>>>(Wk + CDIM * CDIM, Wk, WkdT);

  knn_kernel<<<BATCH * 8, 256, 0, stream>>>(qpos, idx, scl);

  dim3 g1((unsigned)(NR / BM), CDIM / BN);
  gemm384_bf16_kernel<<<g1, 256, 0, stream>>>(qbf, nullptr, 1, 0, WvT,  bv, nullptr, voff);
  gemm384_bf16_kernel<<<g1, 256, 0, stream>>>(qbf, nullptr, 1, 0, W1bT, b1, qW1b, nullptr);
  gemm384_bf16_kernel<<<g1, 256, 0, stream>>>(qbf, nullptr, 1, 0, WkdT, bk, qWkd, nullptr);

  dim3 g2((unsigned)(M2 / BM), CDIM / BN);
  // G = gather(voff, idx) @ W1_top   (row-gathered A)
  gemm384_bf16_kernel<<<g2, 256, 0, stream>>>(voff, idx, NPTS * KNN, NPTS, W1tT, nullptr, G, nullptr);

  ln_gelu_offset_kernel<<<(int)(M2 / 8), 256, 0, stream>>>(G, qW1b, ln_g, ln_b, W2, qpos, idx, scl, spos);
  three_nn_kernel<<<BATCH * 80, 256, 0, stream>>>(qpos, spos, idx3, w3);
  interp_kernel<<<(int)(M2 / 8), 256, 0, stream>>>(q, idx3, w3, A2);

  // P(=G) = interp @ Wk_top ; final: max_k leaky_relu(P + q@(Wk_bot - Wk_top) + bk)
  gemm384_bf16_kernel<<<g2, 256, 0, stream>>>(A2, nullptr, 1, 0, WktT, nullptr, G, nullptr);
  reduce_max_kernel<<<(int)((NR * CDIM) / 256), 256, 0, stream>>>(G, qWkd, out);
}
